// UResNet_50800873177490
// MI455X (gfx1250) — matrix operand order repaired and checked by reading, not compiled, not run on hardware
//
#include <hip/hip_runtime.h>
#include <cstdint>
#include <cstddef>

typedef _Float16 half_t;
typedef __attribute__((ext_vector_type(16))) _Float16 v16h;
typedef __attribute__((ext_vector_type(8)))  _Float16 v8h;
typedef __attribute__((ext_vector_type(8)))  float    v8f;

// ---------------------------------------------------------------------------
// Elementwise / setup kernels
// ---------------------------------------------------------------------------
__global__ void fill_f32(float* __restrict__ p, float v, long n){
  long i = (long)blockIdx.x*blockDim.x + threadIdx.x;
  if(i<n) p[i]=v;
}
// convert + transpose + zero-pad conv weights: w[(tap*Cin+ci)*Cout+co] -> o[co*Kpad+kk]
__global__ void cvt_wT(const float* __restrict__ w, half_t* __restrict__ o,
                       int Ktot, int Kpad, int Cout){
  int i = blockIdx.x*blockDim.x + threadIdx.x;
  int tot = Cout*Kpad; if(i>=tot) return;
  int kk = i % Kpad; int co = i / Kpad;
  o[i] = (kk < Ktot) ? (half_t)w[(long)kk*Cout + co] : (half_t)0.f;
}
__global__ void scatter_kernel(const int* __restrict__ c, const float* __restrict__ f,
                               float* __restrict__ grid, float* __restrict__ occ, int n){
  int p = blockIdx.x*blockDim.x + threadIdx.x;
  if(p>=n) return;
  long v = ((long)c[3*p]*64 + c[3*p+1])*64 + c[3*p+2];
  atomicAdd(&grid[v], f[p]);
  atomicAdd(&occ[v], 1.0f);
}
__global__ void mask_from_occ(const float* __restrict__ occ, float* __restrict__ m, long n){
  long i = (long)blockIdx.x*blockDim.x + threadIdx.x;
  if(i<n) m[i] = occ[i] > 0.0f ? 1.0f : 0.0f;
}
__global__ void maxpool2(const float* __restrict__ in, float* __restrict__ out, int Do){
  long i = (long)blockIdx.x*blockDim.x + threadIdx.x;
  long tot = (long)Do*Do*Do; if(i>=tot) return;
  int x = (int)(i % Do); int y = (int)((i/Do)%Do); int z = (int)(i/((long)Do*Do));
  int Di = Do*2; float m = 0.f;
  for(int d=0; d<8; ++d){
    int dz=d>>2, dy=(d>>1)&1, dx=d&1;
    m = fmaxf(m, in[((long)(2*z+dz)*Di + (2*y+dy))*Di + (2*x+dx)]);
  }
  out[i]=m;
}
__global__ void add_ip(float* __restrict__ o, const float* __restrict__ a, long n){
  long i = (long)blockIdx.x*blockDim.x + threadIdx.x;
  if(i<n) o[i] += a[i];
}
__global__ void concat2(const float* __restrict__ a, const float* __restrict__ b,
                        float* __restrict__ o, long V, int C){
  long i = (long)blockIdx.x*blockDim.x + threadIdx.x;
  long tot = V*(long)(2*C); if(i>=tot) return;
  long v = i/(2*C); int c = (int)(i%(2*C));
  o[i] = (c<C) ? a[v*C+c] : b[v*C + (c-C)];
}

// ---------------------------------------------------------------------------
// Masked batch-norm: sufficient stats then apply (+ReLU +mask)
// stats[0]=cnt, stats[1+c]=sum_c, stats[1+C+c]=sumsq_c
// ---------------------------------------------------------------------------
__global__ void bn_stats(const float* __restrict__ x, const float* __restrict__ mask,
                         float* __restrict__ stats, long V, int C){
  int c = blockIdx.y;
  __shared__ float s1[256], s2[256], s3[256];
  float a1=0.f, a2=0.f, a3=0.f;
  for(long v = (long)blockIdx.x*blockDim.x + threadIdx.x; v < V;
      v += (long)gridDim.x*blockDim.x){
    float m  = mask[v];
    float xv = x[v*C + c];
    a1 += xv*m; a2 += xv*xv*m; a3 += m;
  }
  s1[threadIdx.x]=a1; s2[threadIdx.x]=a2; s3[threadIdx.x]=a3;
  __syncthreads();
  for(int s=128; s>0; s>>=1){
    if((int)threadIdx.x < s){
      s1[threadIdx.x]+=s1[threadIdx.x+s];
      s2[threadIdx.x]+=s2[threadIdx.x+s];
      s3[threadIdx.x]+=s3[threadIdx.x+s];
    }
    __syncthreads();
  }
  if(threadIdx.x==0){
    atomicAdd(&stats[1+c],   s1[0]);
    atomicAdd(&stats[1+C+c], s2[0]);
    if(c==0) atomicAdd(&stats[0], s3[0]);
  }
}
__global__ void bn_apply(const float* __restrict__ x, float* __restrict__ y,
                         const float* __restrict__ mask, const float* __restrict__ stats,
                         const float* __restrict__ w, const float* __restrict__ b,
                         long V, int C){
  long i = (long)blockIdx.x*blockDim.x + threadIdx.x;
  if(i >= V*(long)C) return;
  long v = i / C; int c = (int)(i % C);
  float cnt  = stats[0];
  float mean = stats[1+c] / cnt;
  float var  = stats[1+C+c] / cnt - mean*mean;
  float yv = (x[i]-mean)*rsqrtf(var + 1e-4f)*w[c] + b[c];
  y[i] = fmaxf(yv, 0.f) * mask[v];
}

// ---------------------------------------------------------------------------
// 3x3x3 SAME conv as implicit GEMM on WMMA (f16 in, f32 acc).
// Stage 1: build a 16 x Kpad im2col patch matrix in LDS (zero padded).
// Stage 2: K-loop where, with contiguous k = 16h + s per lane half, the A fragment
//          is two contiguous ds_load_b128 and the B fragment (weights
//          pre-transposed to [Cout][Kpad]) is two contiguous global b128.
// One wave = 16-voxel tile x 16 out channels; waves share the LDS im2col.
// ---------------------------------------------------------------------------
__global__ void conv3_wmma(const float* __restrict__ in, const half_t* __restrict__ wtT,
                           const float* __restrict__ mask, float* __restrict__ out,
                           int D, int Cin, int Cout, int Kpad){
  extern __shared__ half_t ldsA[];          // [16][Kpad]
  const int xs = (D >= 16) ? 4 : 3;         // log2(XT)
  const int XT = 1 << xs;                   // x strip length (16 or 8)
  const int YT = 16 >> xs;                  // y rows per tile (1 or 2)
  const int lane = threadIdx.x & 31;
  const int wave = threadIdx.x >> 5;        // out-channel tile index
  const int ntx = D >> xs, nty = D / YT;
  int t  = blockIdx.x;
  int xt = t % ntx; t /= ntx;
  int yt = t % nty; t /= nty;
  int z  = t;
  int x0 = xt << xs, y0 = yt * YT;

  const int Ktot = 27*Cin;
  // ---- build im2col: one (tap,ci) decode per K index, reused for 16 voxels
  for (int kk = threadIdx.x; kk < Kpad; kk += blockDim.x) {
    int tap = 0, ci = 0;
    bool valid = (kk < Ktot);
    if (valid) { tap = kk / Cin; ci = kk - tap*Cin; }
    int dz = tap/9, rem = tap - dz*9, dy = rem/3, dx = rem - dy*3;
    int gz = z + dz - 1;
    bool zok = valid && (gz >= 0) && (gz < D);
#pragma unroll
    for (int m = 0; m < 16; ++m) {
      int my = m >> xs, mx = m & (XT-1);
      int gy = y0 + my + dy - 1, gx = x0 + mx + dx - 1;
      float v = 0.f;
      if (zok && gy>=0 && gy<D && gx>=0 && gx<D)
        v = in[(((long)gz*D + gy)*D + gx)*Cin + ci];
      ldsA[m*Kpad + kk] = (half_t)v;
    }
  }
  __syncthreads();

  const int m    = lane & 15;
  const int half = lane >> 4;
  const int n    = (wave<<4) + m;
  const half_t* __restrict__ arow = ldsA + m*Kpad + 16*half;  // A: this lane's K base
  const half_t* __restrict__ brow = wtT + (long)n*Kpad + 16*half; // B: this lane's K base
  v8f acc = {};
  for (int kb = 0; kb < Kpad; kb += 32) {
    __builtin_prefetch(brow + kb + 32, 0, 0);
    v8h a_lo = *(const v8h*)(arow + kb);        // K = kb+16h .. +7  -> a[0..7]
    v8h a_hi = *(const v8h*)(arow + kb + 8);    // K = kb+16h+8 ..   -> a[8..15]
    v8h b_lo = *(const v8h*)(brow + kb);        // K = kb+16h .. +7  -> b[0..7]
    v8h b_hi = *(const v8h*)(brow + kb + 8);    // K = kb+16h+8 ..   -> b[8..15]
    v16h a, bm;
#pragma unroll
    for (int e = 0; e < 8; ++e) {
      a[e] = a_lo[e];  a[8+e]  = a_hi[e];
      bm[e] = b_lo[e]; bm[8+e] = b_hi[e];
    }
    acc = __builtin_amdgcn_wmma_f32_16x16x32_f16(
        /*neg_a=*/false, a, /*neg_b=*/false, bm,
        /*c_mod=*/(short)0, acc, /*reuse_a=*/false, /*reuse_b=*/false);
  }
  // D fragment: VGPR r -> row r + 8*half, col = lane&15
#pragma unroll
  for (int r = 0; r < 8; ++r) {
    int mm = r + 8*half;
    int oy = y0 + (mm >> xs), ox = x0 + (mm & (XT-1));
    long v = ((long)z*D + oy)*D + ox;
    out[v*Cout + n] = acc[r] * mask[v];
  }
}

// ---------------------------------------------------------------------------
// 2x2x2 stride-2 down conv (VALID), transpose up conv, 1x1 nin — plain VALU
// (these are <5% of total FLOPs)
// ---------------------------------------------------------------------------
__global__ void conv_down(const float* __restrict__ in, const float* __restrict__ w,
                          const float* __restrict__ mask, float* __restrict__ out,
                          int Do, int Cin, int Cout){
  long i = (long)blockIdx.x*blockDim.x + threadIdx.x;
  long tot = (long)Do*Do*Do*Cout; if(i>=tot) return;
  int co = (int)(i % Cout); long v = i / Cout;
  int x = (int)(v % Do); int y = (int)((v/Do)%Do); int z = (int)(v/((long)Do*Do));
  int Di = 2*Do; float s = 0.f;
  for(int tap=0; tap<8; ++tap){
    int dz=tap>>2, dy=(tap>>1)&1, dx=tap&1;
    long iv = (((long)(2*z+dz)*Di + (2*y+dy))*Di + (2*x+dx))*Cin;
    const float* wp = w + (long)tap*Cin*Cout + co;
    for(int ci=0; ci<Cin; ++ci) s += in[iv+ci]*wp[(long)ci*Cout];
  }
  out[i] = s * mask[v];
}
__global__ void conv_up(const float* __restrict__ in, const float* __restrict__ w,
                        const float* __restrict__ mask, float* __restrict__ out,
                        int Do, int Cin, int Cout){
  long i = (long)blockIdx.x*blockDim.x + threadIdx.x;
  long tot = (long)Do*Do*Do*Cout; if(i>=tot) return;
  int co = (int)(i % Cout); long v = i / Cout;
  int x = (int)(v % Do); int y = (int)((v/Do)%Do); int z = (int)(v/((long)Do*Do));
  int Di = Do/2;
  int tap = ((z&1)<<2) | ((y&1)<<1) | (x&1);
  long iv = (((long)(z>>1)*Di + (y>>1))*Di + (x>>1))*Cin;
  const float* wp = w + (long)tap*Cin*Cout + co;
  float s = 0.f;
  for(int ci=0; ci<Cin; ++ci) s += in[iv+ci]*wp[(long)ci*Cout];
  out[i] = s * mask[v];
}
__global__ void nin_mm(const float* __restrict__ in, const float* __restrict__ w,
                       float* __restrict__ out, long V, int Cin, int Cout){
  long i = (long)blockIdx.x*blockDim.x + threadIdx.x;
  long tot = V*(long)Cout; if(i>=tot) return;
  int co = (int)(i % Cout); long v = i / Cout;
  float s = 0.f;
  for(int ci=0; ci<Cin; ++ci) s += in[v*Cin+ci]*w[(long)ci*Cout+co];
  out[i] = s;
}
// gather features at point coords, apply 16x5 linear head
__global__ void head_kernel(const int* __restrict__ coords, const float* __restrict__ x,
                            const float* __restrict__ lw, const float* __restrict__ lb,
                            float* __restrict__ out, int n){
  int p = blockIdx.x*blockDim.x + threadIdx.x;
  if(p>=n) return;
  long base = (((long)coords[3*p]*64 + coords[3*p+1])*64 + coords[3*p+2])*16;
  for(int nc=0; nc<5; ++nc){
    float s = lb[nc];
    for(int c=0; c<16; ++c) s += x[base+c]*lw[c*5+nc];
    out[p*5+nc] = s;
  }
}

// ---------------------------------------------------------------------------
// Host-side parameter-offset walk (jax pytree flatten order: sorted dict keys,
// tuples (w,b) in order, lists in order).
// ---------------------------------------------------------------------------
struct Bn  { int w, b; };
struct Blk { Bn bn1, bn2; int c1, c2, nin; int a, bch; };
struct Lvl { bool inner; Bn dbn; int dconv; Blk post[2]; Blk pre[2]; Bn ubn; int uconv; };

struct Walk {
  int off = 0;
  Bn bn(int c){ Bn r{off, off+c}; off += 2*c; return r; }
  Blk blk(int a, int b){
    Blk k; k.a=a; k.bch=b;
    k.bn1 = bn(a); k.bn2 = bn(b);           // keys: bn1 < bn2 < conv1 < conv2 < nin
    k.c1 = off; off += 27*a*b;
    k.c2 = off; off += 27*b*b;
    k.nin = -1; if(a!=b){ k.nin = off; off += a*b; }
    return k;
  }
  void lvl(Lvl* L, const int* planes, int nleft){
    int p0 = planes[0];
    if(nleft > 1){
      int p1 = planes[1];
      L[0].inner = true;                    // keys: down_bn<down_conv<inner<post<pre<up_bn<up_conv
      L[0].dbn  = bn(p0);
      L[0].dconv = off; off += 8*p0*p1;
      lvl(L+1, planes+1, nleft-1);
      L[0].post[0] = blk(2*p0, p0);
      L[0].post[1] = blk(p0, p0);
      L[0].pre[0]  = blk(p0, p0);
      L[0].pre[1]  = blk(p0, p0);
      L[0].ubn  = bn(p1);
      L[0].uconv = off; off += 8*p1*p0;
    } else {
      L[0].inner = false;                   // only 'pre_blocks'
      L[0].pre[0] = blk(p0, p0);
      L[0].pre[1] = blk(p0, p0);
    }
  }
};

struct Ctx {
  hipStream_t stream;
  const float* P;
  float* stats; half_t* wbf;
  float* mask[4];
  float *xA[4], *xB[4], *t1[4], *t2[4], *h[4], *cat[4];
  Lvl L[4];
  int D[4]; int C[4]; long V[4];
};

static inline int nblk(long n){ return (int)((n + 255) / 256); }

static void run_bnrelu(Ctx& c, const float* x, float* y, const float* mask,
                       Bn bn, long V, int C){
  fill_f32<<<1, 256, 0, c.stream>>>(c.stats, 0.f, (long)(1+2*C));
  bn_stats<<<dim3(64, C), 256, 0, c.stream>>>(x, mask, c.stats, V, C);
  bn_apply<<<nblk(V*C), 256, 0, c.stream>>>(x, y, mask, c.stats, c.P+bn.w, c.P+bn.b, V, C);
}
static void run_conv3(Ctx& c, const float* x, int woff, const float* mask, float* y,
                      int D, int Cin, int Cout){
  int Ktot = 27*Cin;
  int Kpad = (Ktot + 31) & ~31;
  cvt_wT<<<nblk((long)Cout*Kpad), 256, 0, c.stream>>>(c.P + woff, c.wbf, Ktot, Kpad, Cout);
  size_t shm = (size_t)16*Kpad*sizeof(half_t);
  int blocks = D*D*D/16;
  conv3_wmma<<<blocks, 32*(Cout/16), shm, c.stream>>>(x, c.wbf, mask, y, D, Cin, Cout, Kpad);
}
static void run_block(Ctx& c, const float* xin, float* xout, float* t1, float* t2,
                      const float* mask, const Blk& bk, long V, int D){
  run_bnrelu(c, xin, t1, mask, bk.bn1, V, bk.a);           // t1 = bnrelu(x)   [a ch]
  run_conv3(c, t1, bk.c1, mask, t2, D, bk.a, bk.bch);      // t2 = conv1(t1)   [b ch]
  run_bnrelu(c, t2, t2, mask, bk.bn2, V, bk.bch);          // in-place
  run_conv3(c, t2, bk.c2, mask, xout, D, bk.bch, bk.bch);  // xout = conv2
  if(bk.nin >= 0){
    nin_mm<<<nblk(V*bk.bch), 256, 0, c.stream>>>(xin, c.P + bk.nin, t1, V, bk.a, bk.bch);
    add_ip<<<nblk(V*bk.bch), 256, 0, c.stream>>>(xout, t1, V*(long)bk.bch);
  } else {
    add_ip<<<nblk(V*bk.bch), 256, 0, c.stream>>>(xout, xin, V*(long)bk.bch);
  }
}
// input in xA[l]; result left in xA[l]
static void run_unet(Ctx& c, int l){
  long V = c.V[l]; int D = c.D[l]; int C = c.C[l];
  const Lvl& L = c.L[l];
  run_block(c, c.xA[l], c.xB[l], c.t1[l], c.t2[l], c.mask[l], L.pre[0], V, D);
  run_block(c, c.xB[l], c.xA[l], c.t1[l], c.t2[l], c.mask[l], L.pre[1], V, D);
  if(L.inner){
    long V1 = c.V[l+1]; int D1 = c.D[l+1]; int C1 = c.C[l+1];
    run_bnrelu(c, c.xA[l], c.t2[l], c.mask[l], L.dbn, V, C);
    conv_down<<<nblk(V1*C1), 256, 0, c.stream>>>(c.t2[l], c.P + L.dconv, c.mask[l+1],
                                                 c.xA[l+1], D1, C, C1);
    run_unet(c, l+1);                                       // result in xA[l+1]
    run_bnrelu(c, c.xA[l+1], c.xA[l+1], c.mask[l+1], L.ubn, V1, C1);
    conv_up<<<nblk(V*C), 256, 0, c.stream>>>(c.xA[l+1], c.P + L.uconv, c.mask[l],
                                             c.h[l], D, C1, C);
    concat2<<<nblk(V*2*C), 256, 0, c.stream>>>(c.xA[l], c.h[l], c.cat[l], V, C);
    run_block(c, c.cat[l], c.xB[l], c.t1[l], c.t2[l], c.mask[l], L.post[0], V, D);
    run_block(c, c.xB[l], c.xA[l], c.t1[l], c.t2[l], c.mask[l], L.post[1], V, D);
  }
}

// ---------------------------------------------------------------------------
extern "C" void kernel_launch(void* const* d_in, const int* in_sizes, int n_in,
                              void* d_out, int out_size, void* d_ws, size_t ws_size,
                              hipStream_t stream){
  const int*   coords = (const int*)d_in[0];
  const float* feats  = (const float*)d_in[1];
  const float* P      = (const float*)d_in[2];
  float* out = (float*)d_out;
  int npts = in_sizes[0] / 3;

  // --- parameter offsets (sorted-key flatten: final_bn, linear, stem, unet) ---
  Walk wk;
  Bn fbn = wk.bn(16);               // final_bn (w,b)
  int lin_b = wk.off; wk.off += 5;  // linear: b then w
  int lin_w = wk.off; wk.off += 80;
  int stem  = wk.off; wk.off += 27*1*16;
  Ctx c; c.stream = stream; c.P = P;
  static const int planes[4] = {16, 32, 48, 64};
  wk.lvl(c.L, planes, 4);

  // --- workspace layout ---
  char* ws = (char*)d_ws;
  size_t o = 0;
  auto alloc = [&](size_t bytes)->char*{
    char* r = ws + o; o += (bytes + 255) & ~(size_t)255; return r;
  };
  const long V0 = 64L*64*64;
  float* grid  = (float*)alloc(V0*4);
  float* occ   = (float*)alloc(V0*4);
  int Dl[4] = {64,32,16,8};
  for(int l=0;l<4;l++){
    c.D[l]=Dl[l]; c.C[l]=planes[l]; c.V[l]=(long)Dl[l]*Dl[l]*Dl[l];
    c.mask[l] = (float*)alloc(c.V[l]*4);
  }
  c.stats = (float*)alloc(512*4);
  c.wbf   = (half_t*)alloc(131072*2);   // >= 48*2592 f16 transposed weights
  for(int l=0;l<4;l++){
    long VC = c.V[l]*(long)c.C[l];
    c.xA[l] = (float*)alloc(VC*4);
    c.xB[l] = (float*)alloc(VC*4);
    c.t2[l] = (float*)alloc(VC*4);
    c.t1[l] = (float*)alloc(VC*2*4);    // up to 2C channels (post-block bn1 on concat)
    if(l<3){
      c.h[l]   = (float*)alloc(VC*4);
      c.cat[l] = (float*)alloc(VC*2*4);
    } else { c.h[l]=nullptr; c.cat[l]=nullptr; }
  }
  (void)ws_size; (void)n_in; (void)out_size;

  // --- 1. scatter points into grid + occupancy ---
  fill_f32<<<nblk(V0), 256, 0, stream>>>(grid, 0.f, V0);
  fill_f32<<<nblk(V0), 256, 0, stream>>>(occ,  0.f, V0);
  scatter_kernel<<<nblk(npts), 256, 0, stream>>>(coords, feats, grid, occ, npts);

  // --- 2. masks at 4 strides ---
  mask_from_occ<<<nblk(V0), 256, 0, stream>>>(occ, c.mask[0], V0);
  for(int l=1;l<4;l++)
    maxpool2<<<nblk(c.V[l]), 256, 0, stream>>>(c.mask[l-1], c.mask[l], c.D[l]);

  // --- 3. stem conv (1 -> 16) ---
  run_conv3(c, grid, stem, c.mask[0], c.xA[0], 64, 1, 16);

  // --- 4. U-Net ---
  run_unet(c, 0);

  // --- 5. final BN + head ---
  run_bnrelu(c, c.xA[0], c.xA[0], c.mask[0], fbn, V0, 16);
  head_kernel<<<nblk(npts), 256, 0, stream>>>(coords, c.xA[0], P + lin_w, P + lin_b,
                                              out, npts);
}
